// LSTM_83030307766642
// MI455X (gfx1250) — matrix operand order repaired and checked by reading, not compiled, not run on hardware
//
#include <hip/hip_runtime.h>
#include <hip/hip_bf16.h>
#include <stdint.h>

// ---------------------------------------------------------------------------
// LSTM on MI455X (gfx1250): bf16 WMMA everywhere.
//   emb   = x @ E^T                 (WMMA GEMM, fp32->bf16 while staging)
//   gates = emb @ [Wf;Wg;Wi;Wo]^T   (4 strip GEMMs into one [8192,1024] buffer)
//   scan  = single persistent workgroup; per-step [16x256]x[256x1024] WMMA,
//           gx slice DMA'd to LDS with global_load_async_to_lds_b128 and
//           overlapped with the WMMAs (ASYNCcnt)
//   out   = G @ Wout^T              (g-projection hoisted out of the scan)
// ---------------------------------------------------------------------------

typedef __attribute__((ext_vector_type(16))) __bf16          v16bf;
typedef __attribute__((ext_vector_type(8)))  float           v8f;
typedef __attribute__((ext_vector_type(16))) unsigned short  v16u16;
typedef __attribute__((ext_vector_type(4)))  unsigned short  v4u16;
typedef __attribute__((ext_vector_type(2)))  unsigned short  v2u16;

#define CV 5000
#define CD 300
#define CH 256
#define CB 16
#define CT 512
#define CBT (CB * CT)   // 8192

__device__ __forceinline__ unsigned short f2bf(float f) {
  // round-to-nearest-even fp32 -> bf16 (software fallback)
  unsigned int u = __builtin_bit_cast(unsigned int, f);
  unsigned int lsb = (u >> 16) & 1u;
  return (unsigned short)((u + 0x7FFFu + lsb) >> 16);
}

// packed fp32x2 -> bf16x2: single v_cvt_pk_bf16_f32 when available
__device__ __forceinline__ v2u16 f2bf2(float lo, float hi) {
#if __has_builtin(__builtin_amdgcn_cvt_pk_bf16_f32)
  auto p = __builtin_amdgcn_cvt_pk_bf16_f32(lo, hi);
  return __builtin_bit_cast(v2u16, p);
#else
  v2u16 r;
  r.x = f2bf(lo);
  r.y = f2bf(hi);
  return r;
#endif
}

__device__ __forceinline__ float sigmoidf_(float x) {
  return 1.0f / (1.0f + __expf(-x));
}

// CDNA5 async DMA: copy 16B from global (saddr + 32-bit voffset) into LDS at
// byte offset lds_off. Tracked by ASYNCcnt.
__device__ __forceinline__ void async_load_b128(unsigned lds_off,
                                                unsigned long long gbase,
                                                unsigned goff) {
  asm volatile("global_load_async_to_lds_b128 %0, %1, %2"
               :: "v"(lds_off), "v"(goff), "s"(gbase)
               : "memory");
}
__device__ __forceinline__ void wait_asynccnt0() {
  asm volatile("s_wait_asynccnt 0x0" ::: "memory");
}

// ---------------------------------------------------------------------------
// fp32 -> bf16 cast (used once for the recurrent weights Uh = [Uf;Ug;Ui;Uo])
// ---------------------------------------------------------------------------
__global__ void cast_f32_bf16(const float* __restrict__ src,
                              unsigned short* __restrict__ dst, int n) {
  int i = blockIdx.x * blockDim.x + threadIdx.x;
  if (i < n) dst[i] = f2bf(src[i]);
}

// ---------------------------------------------------------------------------
// Generic GEMM:  C[M,N] = A[M,K] * Bt[N,K]^T   (A,Bt fp32 row-major, lda=ldb=K)
// 128x128 tile / 256-thread block (8 waves, wave32). LDS holds bf16 fragments
// in fragment-major layout (lane half h holds k = 16h..16h+15), so each lane reads its 32B fragment
// contiguously. Interior tiles: unguarded float4 loads + packed cvt +
// ds_store_b64.
// ---------------------------------------------------------------------------
__global__ void __launch_bounds__(256)
gemm_bf16_wmma(const float* __restrict__ A, const float* __restrict__ Bt,
               float* __restrict__ C, int M, int N, int K, int ldc) {
  __shared__ __align__(32) unsigned short AsF[8][32][16];   // 8 KB
  __shared__ __align__(32) unsigned short BsF[8][32][16];   // 8 KB

  const int tid  = threadIdx.x;
  const int lane = tid & 31;
  const int wave = tid >> 5;
  const int wr   = wave >> 1;   // 0..3 -> 32-row group
  const int wc   = wave & 1;    // 0..1 -> 64-col group
  const int m0   = blockIdx.y * 128;
  const int n0   = blockIdx.x * 128;

  const bool tileIn = (m0 + 128 <= M) && (n0 + 128 <= N);

  v8f acc[2][4] = {};

  for (int k0 = 0; k0 < K; k0 += 32) {
    __syncthreads();
    if (tileIn && (k0 + 32 <= K)) {
      // ---------------- fast path: vectorized, unguarded ----------------
      #pragma unroll
      for (int it = 0; it < 4; ++it) {
        int slot = tid + it * 256;      // 0..1023, one float4 each
        int row  = slot >> 3;           // 0..127
        int c4   = slot & 7;            // float4 index within the 32 k's
        {
          float4 v = *(const float4*)(A + (size_t)(m0 + row) * K + k0 + c4 * 4);
          v2u16 lo = f2bf2(v.x, v.y);
          v2u16 hi = f2bf2(v.z, v.w);
          v4u16 us = {lo.x, lo.y, hi.x, hi.y};
          int lg = (c4 >> 2) & 1;                       // A: k-half by (k>>4)&1
          int eb = (c4 & 3) * 4;                        // elem base in frag
          *(v4u16*)&AsF[row >> 4][lg * 16 + (row & 15)][eb] = us;
        }
        {
          float4 v = *(const float4*)(Bt + (size_t)(n0 + row) * K + k0 + c4 * 4);
          v2u16 lo = f2bf2(v.x, v.y);
          v2u16 hi = f2bf2(v.z, v.w);
          v4u16 us = {lo.x, lo.y, hi.x, hi.y};
          int lg = (c4 >> 2) & 1;                       // B: k-half by (k>>4)&1
          int eb = (c4 & 3) * 4;
          *(v4u16*)&BsF[row >> 4][lg * 16 + (row & 15)][eb] = us;
        }
      }
      // prefetch next k-tile (speculative; drops silently if OOB)
      if (k0 + 32 < K) {
        int prow = tid & 127;
        __builtin_prefetch(A  + (size_t)(m0 + prow) * K + k0 + 32, 0);
        __builtin_prefetch(Bt + (size_t)(n0 + prow) * K + k0 + 32, 0);
      }
    } else {
      // ---------------- slow path: guarded scalar (edges / K tail) ------
      #pragma unroll
      for (int it = 0; it < 4; ++it) {
        int slot = tid + it * 256;
        int row  = slot >> 3;
        int c4   = slot & 7;
        int gr   = m0 + row;
        int gn   = n0 + row;
        #pragma unroll
        for (int j = 0; j < 4; ++j) {
          int k  = c4 * 4 + j;
          int gk = k0 + k;
          float va = 0.0f, vb = 0.0f;
          if (gr < M && gk < K) va = A [(size_t)gr * K + gk];
          if (gn < N && gk < K) vb = Bt[(size_t)gn * K + gk];
          int lga = (k >> 4) & 1;
          int ea  = k & 15;
          AsF[row >> 4][lga * 16 + (row & 15)][ea] = f2bf(va);
          int lgb = (k >> 4) & 1;
          BsF[row >> 4][lgb * 16 + (row & 15)][k & 15] = f2bf(vb);
        }
      }
    }
    __syncthreads();

    v16bf a[2], b[4];
    #pragma unroll
    for (int mt = 0; mt < 2; ++mt)
      a[mt] = __builtin_bit_cast(v16bf, *(const v16u16*)AsF[wr * 2 + mt][lane]);
    #pragma unroll
    for (int nt = 0; nt < 4; ++nt)
      b[nt] = __builtin_bit_cast(v16bf, *(const v16u16*)BsF[wc * 4 + nt][lane]);

    #pragma unroll
    for (int mt = 0; mt < 2; ++mt)
      #pragma unroll
      for (int nt = 0; nt < 4; ++nt)
        acc[mt][nt] = __builtin_amdgcn_wmma_f32_16x16x32_bf16(
            false, a[mt], false, b[nt], (short)0, acc[mt][nt], false, false);
  }

  // ---- epilogue: C layout is VGPR e -> row (e + 8*(lane>=16)), col = lane&15
  const int chi = (lane >> 4) * 8;
  const int cn  = lane & 15;
  #pragma unroll
  for (int mt = 0; mt < 2; ++mt) {
    #pragma unroll
    for (int nt = 0; nt < 4; ++nt) {
      int col = n0 + wc * 64 + nt * 16 + cn;
      if (col >= N) continue;
      #pragma unroll
      for (int e = 0; e < 8; ++e) {
        int row = m0 + wr * 32 + mt * 16 + chi + e;
        if (row < M) C[(size_t)row * ldc + col] = acc[mt][nt][e];
      }
    }
  }
}

// ---------------------------------------------------------------------------
// Sequential LSTM scan: single 256-thread workgroup, 512 steps.
// Wave w owns gate columns [128w, 128w+128). Per step:
//   z = gx[:, t, :] + h @ Uh^T   via 8x8x8 = 512 WMMAs (M=16, K=256, N=1024)
// gx slice for step t+1 is DMA'd into LDS asynchronously while step t's
// WMMAs run. h, c, z, gx-slice live in LDS (160 KB of the 320 KB WGP budget).
// ---------------------------------------------------------------------------
__global__ void __launch_bounds__(256)
lstm_scan(const float* __restrict__ h0, const float* __restrict__ c0,
          const float* __restrict__ gx,            // [8192, 1024]
          const unsigned short* __restrict__ Uhb,  // [1024, 256] bf16
          float* __restrict__ G) {                 // [8192, 256]
  __shared__ float hS[CB * CH];                    // 16 KB
  __shared__ float cS[CB * CH];                    // 16 KB
  __shared__ float zS[CB * 4 * CH];                // 64 KB
  __shared__ __align__(16) float gxb[CB][4 * CH];  // 64 KB gx slice for step t

  const int tid  = threadIdx.x;
  const int lane = tid & 31;
  const int wave = tid >> 5;
  const int r    = lane & 15;   // batch row in A frag / col in B frag
  const int lg   = lane >> 4;   // lane half

  const unsigned long long gxbase = (unsigned long long)(uintptr_t)gx;

  // broadcast initial state across batch
  for (int i = tid; i < CB * CH; i += 256) {
    hS[i] = h0[i & (CH - 1)];
    cS[i] = c0[i & (CH - 1)];
  }

  // prefill gx slice for t=0 (async DMA into LDS, 4096 x 16B chunks)
  #pragma unroll
  for (int i = 0; i < 16; ++i) {
    int idx = tid + i * 256;
    int b   = idx >> 8;
    int cc4 = (idx & 255) * 4;
    async_load_b128((unsigned)(uintptr_t)&gxb[b][cc4], gxbase,
                    (unsigned)(((b * CT + 0) * 1024 + cc4) * 4));
  }
  wait_asynccnt0();
  __syncthreads();

  for (int t = 0; t < CT; ++t) {
    // ---- bf16 A fragments of h (16 x 256 -> 8 frags of 16x32) ----
    v16bf a[8];
    #pragma unroll
    for (int kf = 0; kf < 8; ++kf) {
      const float* hp = &hS[r * CH + kf * 32 + lg * 16];
      v16u16 u;
      #pragma unroll
      for (int j = 0; j < 8; j += 2) {
        v2u16 p = f2bf2(hp[j], hp[j + 1]);
        u[j] = p.x; u[j + 1] = p.y;
      }
      #pragma unroll
      for (int j = 0; j < 8; j += 2) {
        v2u16 p = f2bf2(hp[8 + j], hp[8 + j + 1]);
        u[8 + j] = p.x; u[9 + j] = p.y;
      }
      a[kf] = __builtin_bit_cast(v16bf, u);
    }

    // ---- accumulator init from the LDS gx slice (C-matrix layout) ----
    v8f acc[8];
    #pragma unroll
    for (int nt = 0; nt < 8; ++nt) {
      int col = wave * 128 + nt * 16 + r;
      #pragma unroll
      for (int e = 0; e < 8; ++e)
        acc[nt][e] = gxb[e + lg * 8][col];
    }
    __syncthreads();   // everyone finished reading gxb for step t

    // ---- kick off async DMA of gx slice for step t+1 (overlaps WMMAs) ----
    if (t + 1 < CT) {
      #pragma unroll
      for (int i = 0; i < 16; ++i) {
        int idx = tid + i * 256;
        int b   = idx >> 8;
        int cc4 = (idx & 255) * 4;
        async_load_b128((unsigned)(uintptr_t)&gxb[b][cc4], gxbase,
                        (unsigned)(((b * CT + (t + 1)) * 1024 + cc4) * 4));
      }
    }

    // ---- z strip for this wave: 8 N-tiles of 16 cols, K = 256 ----
    #pragma unroll 1
    for (int nt = 0; nt < 8; ++nt) {
      const int col = wave * 128 + nt * 16 + r;
      v8f z = acc[nt];
      #pragma unroll
      for (int kf = 0; kf < 8; ++kf) {
        v16bf bfrag = __builtin_bit_cast(
            v16bf, *(const v16u16*)(Uhb + (size_t)col * CH + kf * 32 + lg * 16));
        z = __builtin_amdgcn_wmma_f32_16x16x32_bf16(
            false, a[kf], false, bfrag, (short)0, z, false, false);
      }
      #pragma unroll
      for (int e = 0; e < 8; ++e)
        zS[(e + lg * 8) * 1024 + col] = z[e];
    }
    __syncthreads();

    // ---- fused gate nonlinearity + state update ----
    #pragma unroll
    for (int q = 0; q < 16; ++q) {
      int idx = tid * 16 + q;          // 0..4095
      int b = idx >> 8;
      int h = idx & 255;
      float f = sigmoidf_(zS[b * 1024 + 0 * CH + h]);
      float g = tanhf   (zS[b * 1024 + 1 * CH + h]);
      float i = sigmoidf_(zS[b * 1024 + 2 * CH + h]);
      float o = sigmoidf_(zS[b * 1024 + 3 * CH + h]);
      float c = g * i + cS[b * CH + h] * f;
      cS[b * CH + h] = c;
      hS[b * CH + h] = o * tanhf(c);
      G[(size_t)(b * CT + t) * CH + h] = g;   // faithful: output projects g
    }
    wait_asynccnt0();   // own DMAs done; barrier makes it workgroup-wide
    __syncthreads();
  }
}

// ---------------------------------------------------------------------------
// Host-side orchestration (all on `stream`, graph-capture safe).
// Workspace layout (bytes):
//   emb   @ 0          : 8192*300*4  =  9,830,400
//   gates @ 9830400    : 8192*1024*4 = 33,554,432
//   G     @ 43384832   : 8192*256*4  =  8,388,608
//   Uhbf  @ 51773440   : 1024*256*2  =    524,288   (total ~52.3 MB)
// ---------------------------------------------------------------------------
extern "C" void kernel_launch(void* const* d_in, const int* in_sizes, int n_in,
                              void* d_out, int out_size, void* d_ws, size_t ws_size,
                              hipStream_t stream) {
  const float* x    = (const float*)d_in[0];
  const float* h0   = (const float*)d_in[1];
  const float* c0   = (const float*)d_in[2];
  const float* E    = (const float*)d_in[3];
  const float* Wf   = (const float*)d_in[4];
  const float* Uf   = (const float*)d_in[5];
  const float* Wg   = (const float*)d_in[6];
  const float* Ug   = (const float*)d_in[7];
  const float* Wi   = (const float*)d_in[8];
  const float* Ui   = (const float*)d_in[9];
  const float* Wo   = (const float*)d_in[10];
  const float* Uo   = (const float*)d_in[11];
  const float* Wout = (const float*)d_in[12];
  float* out = (float*)d_out;

  char* ws = (char*)d_ws;
  float*          emb   = (float*)(ws);
  float*          gates = (float*)(ws + 9830400);
  float*          G     = (float*)(ws + 9830400 + 33554432);
  unsigned short* Uhb   = (unsigned short*)(ws + 9830400 + 33554432 + 8388608);

  // cast recurrent weights to bf16, concatenated [Uf;Ug;Ui;Uo] = [1024,256]
  const int nU = CH * CH;
  cast_f32_bf16<<<(nU + 255) / 256, 256, 0, stream>>>(Uf, Uhb + 0 * nU, nU);
  cast_f32_bf16<<<(nU + 255) / 256, 256, 0, stream>>>(Ug, Uhb + 1 * nU, nU);
  cast_f32_bf16<<<(nU + 255) / 256, 256, 0, stream>>>(Ui, Uhb + 2 * nU, nU);
  cast_f32_bf16<<<(nU + 255) / 256, 256, 0, stream>>>(Uo, Uhb + 3 * nU, nU);

  dim3 blk(256);
  // emb[8192,300] = x[8192,5000] @ E[300,5000]^T
  gemm_bf16_wmma<<<dim3((CD + 127) / 128, CBT / 128), blk, 0, stream>>>(
      x, E, emb, CBT, CD, CV, CD);
  // gates[8192,1024] strips = emb @ {Wf,Wg,Wi,Wo}^T
  gemm_bf16_wmma<<<dim3(2, CBT / 128), blk, 0, stream>>>(emb, Wf, gates + 0,   CBT, CH, CD, 1024);
  gemm_bf16_wmma<<<dim3(2, CBT / 128), blk, 0, stream>>>(emb, Wg, gates + 256, CBT, CH, CD, 1024);
  gemm_bf16_wmma<<<dim3(2, CBT / 128), blk, 0, stream>>>(emb, Wi, gates + 512, CBT, CH, CD, 1024);
  gemm_bf16_wmma<<<dim3(2, CBT / 128), blk, 0, stream>>>(emb, Wo, gates + 768, CBT, CH, CD, 1024);
  // sequential recurrence (one persistent workgroup)
  lstm_scan<<<1, 256, 0, stream>>>(h0, c0, gates, Uhb, G);
  // out[8192,5000] = G[8192,256] @ Wout[5000,256]^T
  gemm_bf16_wmma<<<dim3((CV + 127) / 128, CBT / 128), blk, 0, stream>>>(
      G, Wout, out, CBT, CV, CH, CV);
}
